// SelfAttentionBlock_33122787787587
// MI455X (gfx1250) — matrix operand order repaired and checked by reading, not compiled, not run on hardware
//
#include <hip/hip_runtime.h>
#include <hip/hip_bf16.h>

// ---------------------------------------------------------------------------
// Self-attention block for MI455X (gfx1250, wave32, WMMA).
//   B=8, C=256, H=W=64, HW=4096.
//   out = x + gamma * softmax((XWq+bq)(XWk+bk)^T / sqrt(C)) (XWv+bv)
// Fully compute-bound (~150 GFLOP vs ~67MB I/O) -> f16 WMMA w/ f32 accum,
// flash-attention (online softmax) so the 4096x4096 score matrix is never
// materialized.
// ---------------------------------------------------------------------------

typedef _Float16 h8  __attribute__((ext_vector_type(8)));
typedef _Float16 h16 __attribute__((ext_vector_type(16)));
typedef float    f8  __attribute__((ext_vector_type(8)));

#define HW_ 4096
#define C_  256

// ds_swizzle xor-butterfly within 16-lane groups (offset: xor[14:10], and[4:0])
#define SWZ_F(v, imm) __int_as_float(__builtin_amdgcn_ds_swizzle(__float_as_int(v), (imm)))

__device__ __forceinline__ h16 ld16x2(const _Float16* p0, const _Float16* p1) {
  h8 a = *(const h8*)p0;
  h8 b = *(const h8*)p1;
  return __builtin_shufflevector(a, b, 0,1,2,3,4,5,6,7,8,9,10,11,12,13,14,15);
}
__device__ __forceinline__ h16 ld16(const _Float16* p) { return ld16x2(p, p + 8); }

__device__ __forceinline__ f8 wmma_f16(h16 a, h16 b, f8 c) {
  // D = A(16x32 f16) x B(32x16 f16) + C(16x16 f32)
  return __builtin_amdgcn_wmma_f32_16x16x32_f16(false, a, false, b, (short)0, c,
                                                false, false);
}

__device__ __forceinline__ float redmax16(float v) {
  v = fmaxf(v, SWZ_F(v, 0x041F));
  v = fmaxf(v, SWZ_F(v, 0x081F));
  v = fmaxf(v, SWZ_F(v, 0x101F));
  v = fmaxf(v, SWZ_F(v, 0x201F));
  return v;
}
__device__ __forceinline__ float redsum16(float v) {
  v += SWZ_F(v, 0x041F);
  v += SWZ_F(v, 0x081F);
  v += SWZ_F(v, 0x101F);
  v += SWZ_F(v, 0x201F);
  return v;
}

// ---------------------------------------------------------------------------
// Kernel 1: convert the three 256x256 f32 weight matrices to f16 (one pass).
// ---------------------------------------------------------------------------
__global__ __launch_bounds__(256) void cvt_weights_kernel(
    const float* __restrict__ wq, const float* __restrict__ wk,
    const float* __restrict__ wv, _Float16* __restrict__ dst) {
  int i = blockIdx.x * 256 + threadIdx.x;   // 0 .. 196607
  float v;
  if (i < 65536)       v = wq[i];
  else if (i < 131072) v = wk[i - 65536];
  else                 v = wv[i - 131072];
  dst[i] = (_Float16)v;
}

// ---------------------------------------------------------------------------
// Kernel 2: QKV projection.  Per batch b and matrix m:  Out = W(256x256) * X(256xHW) + bias.
// Grid (HW/64 token blocks, 3 matrices, B).  Block = 128 threads (4 waves).
// Wave w computes c_out tile [w*64, w*64+64) x 64 tokens (16 WMMA tiles).
// Q/K written token-major [HW,C] f16 (Q pre-scaled by 1/sqrt(C)=1/16),
// V written channel-major [C,HW] f16.
// ---------------------------------------------------------------------------
__global__ __launch_bounds__(128) void qkv_proj_kernel(
    const float* __restrict__ x, const _Float16* __restrict__ wh,
    const float* __restrict__ bq, const float* __restrict__ bk,
    const float* __restrict__ bv, _Float16* __restrict__ qo,
    _Float16* __restrict__ ko, _Float16* __restrict__ vto) {
  const int tid  = threadIdx.x;
  const int lane = tid & 31;
  const int w    = tid >> 5;
  const int lrow = lane & 15;
  const int lh   = lane >> 4;
  const int aoff = (lane & 16) ? 8 : 0;    // A-frag K offset (ISA 7.12.2)
  const int boff = (lane & 16) ? 8 : 0;    // B-frag K offset
  const int tb   = blockIdx.x * 64;        // token base
  const int mat  = blockIdx.y;             // 0=Q 1=K 2=V
  const int b    = blockIdx.z;

  const _Float16* W   = wh + mat * 65536;
  const float* bias   = (mat == 0) ? bq : (mat == 1) ? bk : bv;
  const float* xb     = x + (size_t)b * C_ * HW_;
  const int m0w       = w * 64;

  // X tile staged token-major: xs[token][cin], row stride 40 halves (80B,
  // 16B-aligned rows, odd-ish dword stride -> conflict-free frag reads).
  __shared__ __align__(64) _Float16 xs[64 * 40];

  f8 acc[4][4] = {};

  for (int kc = 0; kc < 8; ++kc) {
    const int c0 = kc * 32;
    __syncthreads();
    {
      // thread: cin row kk = tid/4, 16 consecutive tokens; transpose into LDS
      const int kk = tid >> 2, t0 = (tid & 3) * 16;
      const float* src = xb + (size_t)(c0 + kk) * HW_ + tb + t0;
      _Float16* dst = xs + kk;
#pragma unroll
      for (int j = 0; j < 16; ++j) dst[(t0 + j) * 40] = (_Float16)src[j];
    }
    __syncthreads();

    h16 af[4];
#pragma unroll
    for (int mt = 0; mt < 4; ++mt) {
      const _Float16* ab = W + (size_t)(m0w + mt * 16 + lrow) * 256 + c0 + aoff;
      af[mt] = ld16x2(ab, ab + 16);
    }
#pragma unroll
    for (int nt = 0; nt < 4; ++nt) {
      h16 bf = ld16x2(xs + (nt * 16 + lrow) * 40 + boff, xs + (nt * 16 + lrow) * 40 + boff + 16);
#pragma unroll
      for (int mt = 0; mt < 4; ++mt) acc[mt][nt] = wmma_f16(af[mt], bf, acc[mt][nt]);
    }
  }

  // Epilogue: bias, (Q-only) 1/sqrt(C) scale, convert, store.
  const float scale = (mat == 0) ? 0.0625f : 1.0f;
  const size_t bo = (size_t)b * C_ * HW_;
#pragma unroll
  for (int mt = 0; mt < 4; ++mt) {
    float bv8[8];
#pragma unroll
    for (int v = 0; v < 8; ++v) bv8[v] = bias[m0w + mt * 16 + v + 8 * lh];
#pragma unroll
    for (int nt = 0; nt < 4; ++nt) {
      if (mat < 2) {
        // token-major: out[tok*C + cout], rows v,v+1 adjacent -> b32 stores
        _Float16* outp = (mat == 0 ? qo : ko) + bo +
                         (size_t)(tb + nt * 16 + lrow) * C_ + m0w + mt * 16 + 8 * lh;
#pragma unroll
        for (int v = 0; v < 8; v += 2) {
          union { _Float16 h[2]; unsigned u; } pk;
          pk.h[0] = (_Float16)((acc[mt][nt][v]     + bv8[v])     * scale);
          pk.h[1] = (_Float16)((acc[mt][nt][v + 1] + bv8[v + 1]) * scale);
          *(unsigned*)(outp + v) = pk.u;
        }
      } else {
        // channel-major: vt[cout*HW + tok], lanes -> consecutive tokens
        _Float16* outp = vto + bo + (size_t)(m0w + mt * 16 + 8 * lh) * HW_ +
                         tb + nt * 16 + lrow;
#pragma unroll
        for (int v = 0; v < 8; ++v)
          outp[(size_t)v * HW_] = (_Float16)(acc[mt][nt][v] + bv8[v]);
      }
    }
  }
}

// ---------------------------------------------------------------------------
// Kernel 3: flash attention + residual.
// Grid (HW/64, B), block = 128 threads (4 waves); wave owns 16 queries,
// full 256-channel output accumulated as O^T = V^T * P^T  (16 ch-tiles).
// K/V 32-key chunks staged in LDS (padded, conflict-free); online softmax
// with ds_swizzle reductions; P routed through per-wave LDS tile to reach
// the B-fragment layout of the second GEMM.
// ---------------------------------------------------------------------------
__global__ __launch_bounds__(128) void flash_attn_kernel(
    const _Float16* __restrict__ q, const _Float16* __restrict__ k,
    const _Float16* __restrict__ vt, const float* __restrict__ x,
    const float* __restrict__ gamma, float* __restrict__ out) {
  const int tid  = threadIdx.x;
  const int lane = tid & 31;
  const int w    = tid >> 5;
  const int lrow = lane & 15;
  const int lh   = lane >> 4;
  const int aoff = (lane & 16) ? 8 : 0;
  const int boff = (lane & 16) ? 8 : 0;
  const int b    = blockIdx.y;
  const int qb   = blockIdx.x * 64 + w * 16;   // this wave's query base
  const size_t bo = (size_t)b * C_ * HW_;

  const _Float16* qB = q  + bo;
  const _Float16* kB = k  + bo;
  const _Float16* vB = vt + bo;

  // K chunk: [32 keys][256 ch], row stride 264 halves (528B -> 16 banks)
  // V chunk: [256 ch][32 keys], row stride 40 halves  (80B  -> 16 banks)
  __shared__ __align__(16) _Float16 ks[32 * 264];
  __shared__ __align__(16) _Float16 vs[256 * 40];
  __shared__ __align__(64) _Float16 pS[4][16 * 32];  // per-wave P tile
  __shared__ float scS[4][16];                       // per-wave row-stat bcast

  // Preload Q fragments for 16 queries x 256 channels (held in VGPRs).
  h16 qf[8];
#pragma unroll
  for (int kc = 0; kc < 8; ++kc) {
    const _Float16* p = qB + (size_t)(qb + lrow) * C_ + kc * 32 + aoff;
    qf[kc] = ld16x2(p, p + 16);
  }

  f8 o[16] = {};
  float mrow[8], lsum[8];
#pragma unroll
  for (int v = 0; v < 8; ++v) { mrow[v] = -3.0e38f; lsum[v] = 0.0f; }

  for (int mc = 0; mc < HW_ / 32; ++mc) {
    const int m0 = mc * 32;
    __syncthreads();  // previous chunk's LDS reads complete
    {
      const _Float16* kc0 = kB + (size_t)m0 * C_;
#pragma unroll
      for (int j = 0; j < 8; ++j) {          // 1024 h8 units / 128 threads
        int u = tid * 8 + j;
        int key = u >> 5, c8 = (u & 31) * 8;
        *(h8*)(ks + key * 264 + c8) = *(const h8*)(kc0 + key * C_ + c8);
      }
#pragma unroll
      for (int j = 0; j < 8; ++j) {
        int u = tid * 8 + j;
        int c = u >> 2, ko = (u & 3) * 8;
        *(h8*)(vs + c * 40 + ko) = *(const h8*)(vB + (size_t)c * HW_ + m0 + ko);
      }
    }
    __syncthreads();

    // S = Qs * K^T for 16 queries x 32 keys (two 16x16 tiles), K over 256 ch.
    f8 s0 = {}, s1 = {};
#pragma unroll
    for (int kc = 0; kc < 8; ++kc) {
      const _Float16* kb0 = ks + lrow * 264 + kc * 32 + boff;
      s0 = wmma_f16(qf[kc], ld16x2(kb0, kb0 + 16), s0);
      s1 = wmma_f16(qf[kc], ld16x2(kb0 + 16 * 264, kb0 + 16 * 264 + 16), s1);
    }

    // Online softmax (rows = queries v+8*lh, cols = keys across lanes).
    float scv[8];
#pragma unroll
    for (int v = 0; v < 8; ++v) {
      float t  = redmax16(fmaxf(s0[v], s1[v]));
      float nm = fmaxf(mrow[v], t);
      float sc = __expf(mrow[v] - nm);
      mrow[v]  = nm;
      float p0 = __expf(s0[v] - nm);
      float p1 = __expf(s1[v] - nm);
      lsum[v]  = lsum[v] * sc + redsum16(p0 + p1);
      scv[v]   = sc;
      const int row = v + 8 * lh;
      pS[w][row * 32 + lrow]      = (_Float16)p0;
      pS[w][row * 32 + 16 + lrow] = (_Float16)p1;
    }
    if (lrow == 0) {
#pragma unroll
      for (int v = 0; v < 8; ++v) scS[w][8 * lh + v] = scv[v];
    }
    // wave-internal LDS RAW (cross-lane): order explicitly, no barrier needed
    asm volatile("s_wait_dscnt 0" ::: "memory");

    const float scq = scS[w][lrow];  // scale for this lane's query column
#pragma unroll
    for (int t = 0; t < 16; ++t)
#pragma unroll
      for (int v = 0; v < 8; ++v) o[t][v] *= scq;

    // O^T += V^T * P^T : A = V^T chunk (16 ch x 32 keys), B = P^T (32 x 16 q)
    h16 pb = ld16x2(pS[w] + lrow * 32 + boff, pS[w] + lrow * 32 + boff + 16);
#pragma unroll
    for (int t = 0; t < 16; ++t) {
      const _Float16* ap = vs + (t * 16 + lrow) * 40 + aoff;
      o[t] = wmma_f16(ld16x2(ap, ap + 16), pb, o[t]);
    }
  }

  // Normalize by row sums (broadcast to query-indexed lanes via LDS).
  if (lrow == 0) {
#pragma unroll
    for (int v = 0; v < 8; ++v) scS[w][8 * lh + v] = lsum[v];
  }
  asm volatile("s_wait_dscnt 0" ::: "memory");
  const float linv = 1.0f / scS[w][lrow];
  const float g = gamma[0];

  // out[b, ch, q] = x + g * O^T  (lanes 0-15 -> 16 consecutive q: coalesced)
#pragma unroll
  for (int t = 0; t < 16; ++t) {
#pragma unroll
    for (int v = 0; v < 8; ++v) {
      size_t addr = bo + (size_t)(t * 16 + v + 8 * lh) * HW_ + qb + lrow;
      out[addr] = x[addr] + g * (o[t][v] * linv);
    }
  }
}

// ---------------------------------------------------------------------------
extern "C" void kernel_launch(void* const* d_in, const int* in_sizes, int n_in,
                              void* d_out, int out_size, void* d_ws, size_t ws_size,
                              hipStream_t stream) {
  (void)in_sizes; (void)n_in; (void)out_size; (void)ws_size;
  const float* x     = (const float*)d_in[0];
  const float* Wq    = (const float*)d_in[1];
  const float* bq    = (const float*)d_in[2];
  const float* Wk    = (const float*)d_in[3];
  const float* bk    = (const float*)d_in[4];
  const float* Wv    = (const float*)d_in[5];
  const float* bv    = (const float*)d_in[6];
  const float* gamma = (const float*)d_in[7];
  float* out = (float*)d_out;

  // Workspace layout (f16 elements): 3 weight mats + Q + K + V^T  (~50.7 MB)
  _Float16* ws    = (_Float16*)d_ws;
  _Float16* wh    = ws;                       // 3 * 65536
  _Float16* qbuf  = ws + 196608;              // 8*4096*256
  _Float16* kbuf  = qbuf + 8388608;
  _Float16* vtbuf = kbuf + 8388608;

  cvt_weights_kernel<<<768, 256, 0, stream>>>(Wq, Wk, Wv, wh);
  qkv_proj_kernel<<<dim3(HW_ / 64, 3, 8), 128, 0, stream>>>(
      x, wh, bq, bk, bv, qbuf, kbuf, vtbuf);
  flash_attn_kernel<<<dim3(HW_ / 64, 8), 128, 0, stream>>>(
      qbuf, kbuf, vtbuf, x, gamma, out);
}
